// CMG_61014305407658
// MI455X (gfx1250) — matrix operand order repaired and checked by reading, not compiled, not run on hardware
//
#include <hip/hip_runtime.h>
#include <hip/hip_bf16.h>

#define B_SZ   8192
#define IN_DIM 576
#define MOT    512
#define CMD    64
#define HID    1024
#define NE     8
#define ODIM   512

typedef __attribute__((ext_vector_type(16))) __bf16 v16bf;
typedef __attribute__((ext_vector_type(8)))  float  v8f;
typedef __attribute__((ext_vector_type(4)))  int    v4i;

union Frag { uint4 u[2]; v16bf v; __bf16 h[16]; };

// ---- CDNA5 async global->LDS staging (compile-safe probe) ----
// Probe-confirmed (round 2 diagnostics): builtin exists with params
// (addrspace(1) v4i*, addrspace(3) v4i*, imm int offset, imm int cpol).
#if defined(__gfx1250__) && \
    __has_builtin(__builtin_amdgcn_global_load_async_to_lds_b128) && \
    __has_builtin(__builtin_amdgcn_s_wait_asynccnt)
#define USE_ASYNC_LDS 1
#define AS1(p) ((__attribute__((address_space(1))) v4i*)(p))
#define AS3(p) ((__attribute__((address_space(3))) v4i*)(p))
#else
#define USE_ASYNC_LDS 0
#endif

// ---------------- elementwise helpers ----------------

__global__ __launch_bounds__(256) void cast_bf16_v4(const float4* __restrict__ in,
                                                    __bf16* __restrict__ out, int n4) {
  int i = blockIdx.x * 256 + threadIdx.x;
  if (i < n4) {
    float4 f = in[i];
    union { __bf16 h[4]; uint2 u; } t;
    t.h[0] = (__bf16)f.x; t.h[1] = (__bf16)f.y;
    t.h[2] = (__bf16)f.z; t.h[3] = (__bf16)f.w;
    *(uint2*)(out + 4 * (size_t)i) = t.u;
  }
}

__global__ __launch_bounds__(256) void concat_cast(const float* __restrict__ motion,
                                                   const float* __restrict__ command,
                                                   __bf16* __restrict__ out) {
  int idx = blockIdx.x * 256 + threadIdx.x;
  if (idx >= B_SZ * IN_DIM) return;
  int b = idx / IN_DIM, c = idx - b * IN_DIM;
  float v = (c < MOT) ? motion[(size_t)b * MOT + c]
                      : command[(size_t)b * CMD + (c - MOT)];
  out[idx] = (__bf16)v;
}

// gating layer 3 (K=1024, N=8) + softmax, one thread per row
__global__ __launch_bounds__(256) void gate3_softmax(const __bf16* __restrict__ g2,
                                                     const float* __restrict__ w3,
                                                     const float* __restrict__ b3,
                                                     float* __restrict__ coeffs) {
  __shared__ float sw[HID * NE];                       // 32 KB
  for (int i = threadIdx.x; i < HID * NE; i += 256) sw[i] = w3[i];
  __syncthreads();
  int b = blockIdx.x * 256 + threadIdx.x;
  float acc[NE];
#pragma unroll
  for (int e = 0; e < NE; ++e) acc[e] = b3[e];
  const __bf16* xr = g2 + (size_t)b * HID;
  for (int i = 0; i < HID; ++i) {
    float xv = (float)xr[i];
#pragma unroll
    for (int e = 0; e < NE; ++e) acc[e] += xv * sw[i * NE + e];
  }
  float mx = acc[0];
#pragma unroll
  for (int e = 1; e < NE; ++e) mx = fmaxf(mx, acc[e]);
  float s = 0.f;
#pragma unroll
  for (int e = 0; e < NE; ++e) { acc[e] = expf(acc[e] - mx); s += acc[e]; }
  float inv = 1.0f / s;
#pragma unroll
  for (int e = 0; e < NE; ++e) coeffs[(size_t)b * NE + e] = acc[e] * inv;
}

// ---------------- main GEMM / MoE-GEMM ----------------
// C[b,n] = act( sum_e c[b,e]*(A@W_e)[b,n] + sum_e c[b,e]*bias_e[n] )
// Expert blending is folded into the A fragment: frag rows are scaled by
// bf16(c[m,e]) before WMMA, so a single fp32 accumulator integrates over
// both K and experts. coeffs==nullptr -> dense GEMM (E==1, no scaling).
// 256 threads = 8 waves; macro-tile 128x128; each wave a 32x64 block
// (2x4 of 16x16x32 bf16 WMMA). Double-buffered LDS staging via
// GLOBAL_LOAD_ASYNC_TO_LDS_B128 (ASYNCcnt), prefetching one k-step ahead.
// Requires M%128==0, N%128==0, K%32==0 (true for all layers here).

#define MT 128
#define NT 128
#define KT 32
#define SA_STR 40    // 32 + 8 halfword pad
#define SB_STR 136   // 128 + 8 halfword pad

template <bool ELU_ACT, bool OUT_BF16>
__global__ __launch_bounds__(256) void gemm_moe(const __bf16* __restrict__ A,
                                                const __bf16* __restrict__ W,
                                                const float* __restrict__ bias,
                                                const float* __restrict__ coeffs,
                                                void* __restrict__ outp,
                                                int M, int N, int K, int E) {
  __shared__ __align__(16) __bf16 sA[2][MT * SA_STR];   // 20.0 KB
  __shared__ __align__(16) __bf16 sB[2][KT * SB_STR];   // 17.0 KB
  __shared__ float sC[MT * NE];                         // 4 KB coeffs
  __shared__ float sBias[NE * NT];                      // 4 KB bias tile

  const int tid  = threadIdx.x;
  const int lane = tid & 31;
  const int wave = tid >> 5;
  const int m0 = blockIdx.y * MT;
  const int n0 = blockIdx.x * NT;
  const int wm = (wave >> 1) * 32;   // 4 row-groups of 32
  const int wn = (wave & 1) * 64;    // 2 col-groups of 64

  if (coeffs) {
    for (int i = tid; i < MT * NE; i += 256) sC[i] = coeffs[(size_t)m0 * NE + i];
  }
  for (int i = tid; i < E * NT; i += 256) {
    int e = i >> 7, n = i & (NT - 1);
    sBias[i] = bias[(size_t)e * N + n0 + n];
  }

  const int K32 = K / KT;
  const int S   = E * K32;

  // staging maps: A tile 128x32 hw (32B/thread), B tile 32x128 hw (2x16B/thread)
  const int arow  = tid >> 1;
  const int acol  = (tid & 1) * 16;
  const int brow0 = tid >> 4;
  const int brow1 = brow0 + 16;
  const int bcol  = (tid & 15) * 8;
  const int mlane = lane & 15;
  const int kbase = (lane >> 4) * 8;    // ISA A-frag k split (lanes 0-15 vs 16-31)
  const int rbase = (lane >> 4) << 3;   // ISA C/D row split

  auto stage = [&](int e, int k0, int buf) {
    const __bf16* gA = A + (size_t)(m0 + arow) * K + (k0 + acol);
    const __bf16* gB = W + ((size_t)e * K + k0) * N + n0;
    __bf16* lA  = &sA[buf][arow * SA_STR + acol];
    __bf16* lB0 = &sB[buf][brow0 * SB_STR + bcol];
    __bf16* lB1 = &sB[buf][brow1 * SB_STR + bcol];
#if USE_ASYNC_LDS
    __builtin_amdgcn_global_load_async_to_lds_b128(AS1(gA),     AS3(lA),     0, 0);
    __builtin_amdgcn_global_load_async_to_lds_b128(AS1(gA + 8), AS3(lA + 8), 0, 0);
    __builtin_amdgcn_global_load_async_to_lds_b128(AS1(gB + (size_t)brow0 * N + bcol), AS3(lB0), 0, 0);
    __builtin_amdgcn_global_load_async_to_lds_b128(AS1(gB + (size_t)brow1 * N + bcol), AS3(lB1), 0, 0);
#else
    uint4 a0 = *(const uint4*)gA;
    uint4 a1 = *(const uint4*)(gA + 8);
    uint4 b0 = *(const uint4*)(gB + (size_t)brow0 * N + bcol);
    uint4 b1 = *(const uint4*)(gB + (size_t)brow1 * N + bcol);
    *(uint4*)lA = a0; *(uint4*)(lA + 8) = a1;
    *(uint4*)lB0 = b0; *(uint4*)lB1 = b1;
#endif
  };

  v8f accF[2][4] = {};

  int e_cur = 0, j_cur = 0;
  stage(0, 0, 0);
  for (int s = 0; s < S; ++s) {
    const int cur = s & 1;
    int e_nxt = e_cur, j_nxt = j_cur + 1;
    if (j_nxt == K32) { j_nxt = 0; ++e_nxt; }

    __syncthreads();                       // prev compute done: safe to overwrite other buf
    if (s + 1 < S) {
      stage(e_nxt, j_nxt * KT, cur ^ 1);   // prefetch next step
#if USE_ASYNC_LDS
      __builtin_amdgcn_s_wait_asynccnt(4); // step s's 4 copies landed; next 4 in flight
#endif
    } else {
#if USE_ASYNC_LDS
      __builtin_amdgcn_s_wait_asynccnt(0);
#endif
    }
    __syncthreads();                       // step s data visible to all waves

    Frag fb[4];
#pragma unroll
    for (int t = 0; t < 4; ++t) {
      const int nb = wn + t * 16;          // B frag: lane%16 = N col, elements over K (table order)
#pragma unroll
      for (int q = 0; q < 8; ++q) {
        fb[t].h[q]     = sB[cur][(kbase + q) * SB_STR + nb + mlane];
        fb[t].h[q + 8] = sB[cur][(16 + kbase + q) * SB_STR + nb + mlane];
      }
    }
    Frag fa[2];
#pragma unroll
    for (int t = 0; t < 2; ++t) {
      const int r = wm + t * 16 + mlane;   // A frag: lane%16 = M row
      fa[t].u[0] = *(const uint4*)&sA[cur][r * SA_STR + kbase];
      fa[t].u[1] = *(const uint4*)&sA[cur][r * SA_STR + 16 + kbase];
      if (coeffs) {                        // fold expert coeff into A rows
        const float c = sC[r * NE + e_cur];
#pragma unroll
        for (int q = 0; q < 16; ++q) fa[t].h[q] = (__bf16)(c * (float)fa[t].h[q]);
      }
    }
#pragma unroll
    for (int tm = 0; tm < 2; ++tm)
#pragma unroll
      for (int tn = 0; tn < 4; ++tn)
        accF[tm][tn] = __builtin_amdgcn_wmma_f32_16x16x32_bf16(
            false, fa[tm].v, false, fb[tn].v, (short)0, accF[tm][tn], false, false);

    e_cur = e_nxt; j_cur = j_nxt;
  }

  // epilogue: + sum_e c*bias, optional ELU, store bf16 or f32
#pragma unroll
  for (int tm = 0; tm < 2; ++tm) {
#pragma unroll
    for (int tn = 0; tn < 4; ++tn) {
      const int ncol = wn + tn * 16 + mlane;
#pragma unroll
      for (int r = 0; r < 8; ++r) {
        const int rowm = wm + tm * 16 + rbase + r;
        float bb;
        if (coeffs) {
          bb = 0.f;
#pragma unroll
          for (int e = 0; e < NE; ++e) bb += sC[rowm * NE + e] * sBias[e * NT + ncol];
        } else {
          bb = sBias[ncol];
        }
        float v = accF[tm][tn][r] + bb;
        if (ELU_ACT) v = (v > 0.0f) ? v : (expf(v) - 1.0f);
        const size_t row = (size_t)m0 + rowm;
        const size_t col = (size_t)n0 + ncol;
        if (OUT_BF16) ((__bf16*)outp)[row * N + col] = (__bf16)v;
        else          ((float*)outp)[row * N + col]  = v;
      }
    }
  }
}

// ---------------- host orchestration ----------------

extern "C" void kernel_launch(void* const* d_in, const int* in_sizes, int n_in,
                              void* d_out, int out_size, void* d_ws, size_t ws_size,
                              hipStream_t stream) {
  const float* motion  = (const float*)d_in[0];
  const float* command = (const float*)d_in[1];
  const float* g_w1 = (const float*)d_in[2];
  const float* g_b1 = (const float*)d_in[3];
  const float* g_w2 = (const float*)d_in[4];
  const float* g_b2 = (const float*)d_in[5];
  const float* g_w3 = (const float*)d_in[6];
  const float* g_b3 = (const float*)d_in[7];
  const float* w[6]  = {(const float*)d_in[8],  (const float*)d_in[10], (const float*)d_in[12],
                        (const float*)d_in[14], (const float*)d_in[16], (const float*)d_in[18]};
  const float* bs[6] = {(const float*)d_in[9],  (const float*)d_in[11], (const float*)d_in[13],
                        (const float*)d_in[15], (const float*)d_in[17], (const float*)d_in[19]};

  char* p = (char*)d_ws;
  auto carve = [&](size_t elems) {
    void* q = (void*)p;
    p += (elems * 2 + 255) & ~(size_t)255;
    return (__bf16*)q;
  };

  __bf16* gw1b = carve((size_t)IN_DIM * HID);
  __bf16* gw2b = carve((size_t)HID * HID);
  size_t wsz[6] = {(size_t)NE * IN_DIM * HID, (size_t)NE * HID * HID, (size_t)NE * HID * HID,
                   (size_t)NE * HID * HID,    (size_t)NE * HID * HID, (size_t)NE * HID * ODIM};
  __bf16* wb[6];
  for (int i = 0; i < 6; ++i) wb[i] = carve(wsz[i]);
  __bf16* xin  = carve((size_t)B_SZ * IN_DIM);
  __bf16* actA = carve((size_t)B_SZ * HID);
  __bf16* actB = carve((size_t)B_SZ * HID);
  float* coeffs = (float*)p;  // 256-aligned

  auto cast = [&](const float* src, __bf16* dst, size_t n) {
    int n4 = (int)(n / 4);
    cast_bf16_v4<<<(n4 + 255) / 256, 256, 0, stream>>>((const float4*)src, dst, n4);
  };
  cast(g_w1, gw1b, (size_t)IN_DIM * HID);
  cast(g_w2, gw2b, (size_t)HID * HID);
  for (int i = 0; i < 6; ++i) cast(w[i], wb[i], wsz[i]);

  concat_cast<<<(B_SZ * IN_DIM + 255) / 256, 256, 0, stream>>>(motion, command, xin);

  const dim3 blk(256);
  const dim3 gH(HID / NT, B_SZ / MT);    // (8, 64)
  const dim3 gO(ODIM / NT, B_SZ / MT);   // (4, 64)

  // gating: ELU(x@g_w1+b1), ELU(.@g_w2+b2), softmax(.@g_w3+b3)
  gemm_moe<true, true><<<gH, blk, 0, stream>>>(xin,  gw1b, g_b1, nullptr, actA,
                                               B_SZ, HID, IN_DIM, 1);
  gemm_moe<true, true><<<gH, blk, 0, stream>>>(actA, gw2b, g_b2, nullptr, actB,
                                               B_SZ, HID, HID, 1);
  gate3_softmax<<<B_SZ / 256, blk, 0, stream>>>(actB, g_w3, g_b3, coeffs);

  // MoE stack (ELU between all but last)
  gemm_moe<true, true><<<gH, blk, 0, stream>>>(xin,  wb[0], bs[0], coeffs, actA,
                                               B_SZ, HID, IN_DIM, NE);
  gemm_moe<true, true><<<gH, blk, 0, stream>>>(actA, wb[1], bs[1], coeffs, actB,
                                               B_SZ, HID, HID, NE);
  gemm_moe<true, true><<<gH, blk, 0, stream>>>(actB, wb[2], bs[2], coeffs, actA,
                                               B_SZ, HID, HID, NE);
  gemm_moe<true, true><<<gH, blk, 0, stream>>>(actA, wb[3], bs[3], coeffs, actB,
                                               B_SZ, HID, HID, NE);
  gemm_moe<true, true><<<gH, blk, 0, stream>>>(actB, wb[4], bs[4], coeffs, actA,
                                               B_SZ, HID, HID, NE);
  gemm_moe<false, false><<<gO, blk, 0, stream>>>(actA, wb[5], bs[5], coeffs, d_out,
                                                 B_SZ, ODIM, HID, NE);
}
